// FinalModel_1855425872284
// MI455X (gfx1250) — matrix operand order repaired and checked by reading, not compiled, not run on hardware
//
#include <hip/hip_runtime.h>

// ---------------------------------------------------------------------------
// 2-layer GRU (B=64, S=4096, I=16, H=128) + MLP head + cumsum for MI455X.
//
// Strategy: recurrence is latency-bound, not BW-bound. Split batch into 4
// workgroups of M=16 rows (one WMMA tile). Layer-0 + head weights live in
// LDS (bf16, ~230 KB of the 320 KB WGP LDS); layer-1 weights live in
// per-wave VGPR B-fragments (64 regs). Paired LDS buffers are merged so DS
// ops share one address register + 16-bit immediate offsets. x_{t+1} is
// prefetched (global_prefetch_b8) at loop top and staged into a ping-pong
// LDS panel by waves 4-11 during the head phase (waves 0-3 do WMMAs), so
// no value lives across the loop body -> no spills, no exposed load waits.
// 5 barriers/step. HBM traffic = x (16.8MB) + out (1MB).
// ---------------------------------------------------------------------------

#define IN     16
#define H      128
#define BATCH  64
#define SEQ    4096
#define MLPD   64
#define G3     384      // 3*H gate width
#define MROWS  16       // batch rows per workgroup
#define NWAVES 24       // one wave per 16-wide gate column tile
#define NTHREADS (NWAVES * 32)

typedef __attribute__((ext_vector_type(16))) __bf16 v16bf;
typedef __attribute__((ext_vector_type(8)))  __bf16 v8bf;
typedef __attribute__((ext_vector_type(8)))  float  v8f;

union ABFrag { v16bf v; v8bf h[2]; };

#define WMMA_BF16(a, b, c) \
  __builtin_amdgcn_wmma_f32_16x16x32_bf16(false, (a), false, (b), (short)0, (c), false, false)

// Fast transcendentals: CDNA5 has V_TANH_F32 (TRANS32). Sigmoid via tanh:
// sigma(x) = 0.5 + 0.5*tanh(x/2)  -> one TRANS op + one FMA, no IEEE divide.
__device__ __forceinline__ float tanh_fast(float x) {
#if __has_builtin(__builtin_amdgcn_tanhf)
  return __builtin_amdgcn_tanhf(x);
#elif __has_builtin(__builtin_amdgcn_tanh_f32)
  return __builtin_amdgcn_tanh_f32(x);
#else
  // tanh(x) = 1 - 2/(exp(2x)+1); rcp instead of IEEE divide
  return 1.0f - 2.0f * __builtin_amdgcn_rcpf(1.0f + __expf(2.0f * x));
#endif
}
__device__ __forceinline__ float sigmoid_f(float x) {
  return 0.5f + 0.5f * tanh_fast(0.5f * x);
}

// A-fragment (16x32 bf16): lane<16 holds row=lane, K k0..k0+15;
// lane>=16 holds K k0+16..k0+31. Two 16-byte LDS loads per lane.
__device__ __forceinline__ v16bf load_afrag(const __bf16* panel, int stride, int lane, int k0) {
  const int hf = lane >> 4;
  const int r  = lane & 15;
  const __bf16* base = panel + r * stride + k0 + hf * 16;
  ABFrag f;
  f.h[0] = *(const v8bf*)(base);
  f.h[1] = *(const v8bf*)(base + 8);
  return f.v;
}

// B-fragment (32x16 bf16): lane<16 = column lane, K k0..k0+15; lane>=16 = K k0+16..k0+31.
// Weights stored [N][Kstride] row-major (PyTorch layout), so 16 contiguous bf16 per lane.
__device__ __forceinline__ v16bf load_bfrag(const __bf16* w, int stride, int lane, int n0, int k0) {
  const int hf = lane >> 4;
  const int c  = lane & 15;
  const __bf16* base = w + (n0 + c) * stride + k0 + hf * 16;
  ABFrag f;
  f.h[0] = *(const v8bf*)(base);
  f.h[1] = *(const v8bf*)(base + 8);
  return f.v;
}

// Build a B-fragment directly from a global f32 [N][K] row-major weight matrix.
__device__ __forceinline__ v16bf make_bfrag_global(const float* __restrict__ w, int K,
                                                   int col, int k0) {
  ABFrag f;
#pragma unroll
  for (int i = 0; i < 16; ++i) f.v[i] = (__bf16)w[col * K + k0 + i];
  return f.v;
}

// Vectorized GRU cell update: 512 threads x 4 consecutive hidden units.
// float4 (b128) LDS loads for all gate slices, float4 store for the f32
// h-master, packed 4xbf16 (8B) store for the WMMA A-panel.
__device__ __forceinline__ void gru_update(const float* gi_base, const float* gh_base,
                                           float* hst, __bf16* ah, int tid) {
  if (tid < MROWS * H / 4) {
    const int m = tid >> 5;          // batch row
    const int j = (tid & 31) << 2;   // hidden unit group of 4
    const float* gi = gi_base + m * G3 + j;
    const float* gh = gh_base + m * G3 + j;
    float4 gir = *(const float4*)(gi);
    float4 ghr = *(const float4*)(gh);
    float4 giz = *(const float4*)(gi + H);
    float4 ghz = *(const float4*)(gh + H);
    float4 gin = *(const float4*)(gi + 2 * H);
    float4 ghn = *(const float4*)(gh + 2 * H);
    float4 hv  = *(const float4*)(hst + m * H + j);
    const float* pgir = (const float*)&gir; const float* pghr = (const float*)&ghr;
    const float* pgiz = (const float*)&giz; const float* pghz = (const float*)&ghz;
    const float* pgin = (const float*)&gin; const float* pghn = (const float*)&ghn;
    const float* ph   = (const float*)&hv;
    float4 hw; float* phw = (float*)&hw;
    union { __bf16 b[4]; unsigned long long u; } pk;
#pragma unroll
    for (int c = 0; c < 4; ++c) {
      float r  = sigmoid_f(pgir[c] + pghr[c]);
      float z  = sigmoid_f(pgiz[c] + pghz[c]);
      float n  = tanh_fast(pgin[c] + r * pghn[c]);
      float hn = (1.0f - z) * n + z * ph[c];
      phw[c]  = hn;
      pk.b[c] = (__bf16)hn;
    }
    *(float4*)(hst + m * H + j) = hw;
    *(unsigned long long*)(ah + m * H + j) = pk.u;
  }
}

__global__ __launch_bounds__(NTHREADS, 1)
void gru_metabolism_kernel(const float* __restrict__ x,
                           const float* __restrict__ w_ih0, const float* __restrict__ w_hh0,
                           const float* __restrict__ b_ih0, const float* __restrict__ b_hh0,
                           const float* __restrict__ w_ih1, const float* __restrict__ w_hh1,
                           const float* __restrict__ b_ih1, const float* __restrict__ b_hh1,
                           const float* __restrict__ w1,   const float* __restrict__ b1,
                           const float* __restrict__ w2,   const float* __restrict__ b2,
                           const float* __restrict__ init_met,
                           float* __restrict__ out)
{
  // ~230 KB static LDS (320 KB per-WG limit on CDNA5). Paired buffers merged
  // so DS ops share an address register + immediate offset.
  __shared__ __align__(16) __bf16 sWih0[G3][32];     // layer0 input weights (K padded)
  __shared__ __align__(16) __bf16 sWhh0[G3][H];      // layer0 hidden weights
  __shared__ __align__(16) __bf16 sW1h[MLPD][H];     // head w1, h-part
  __shared__ __align__(16) __bf16 sW1x[MLPD][32];    // head w1, x-part (padded)
  __shared__ __align__(16) __bf16 sAxt2[2][MROWS][32]; // ping-pong x_t panels
  __shared__ __align__(16) __bf16 sAh[2][MROWS][H];  // [0]=h0, [1]=h1 bf16 A-panels
  __shared__ __align__(16) float sG[2][MROWS][G3];   // [0]=Gi, [1]=Gh gate pre-acts
  __shared__ __align__(16) float sHdn[MROWS][MLPD];  // head hidden
  __shared__ __align__(16) float sH[2][MROWS][H];    // [0]=h0, [1]=h1 f32 masters
  __shared__ __align__(16) float sBias[4][G3];       // bih0, bhh0, bih1, bhh1
  __shared__ float sB1v[MLPD];
  __shared__ __align__(16) float sW2v[MLPD];

  const int tid  = threadIdx.x;
  const int lane = tid & 31;
  const int wv   = tid >> 5;          // 0..23
  const int l16  = lane & 15;
  const int hf   = lane >> 4;
  const int b0   = blockIdx.x * MROWS;
  const int n0   = wv * 16;           // gate-column tile base for this wave

  // ---------------- one-time init ----------------
  for (int idx = tid; idx < G3 * 32; idx += NTHREADS) {
    int n = idx >> 5, k = idx & 31;
    sWih0[n][k] = (k < IN) ? (__bf16)w_ih0[n * IN + k] : (__bf16)0.0f;
  }
  for (int idx = tid; idx < G3 * H; idx += NTHREADS) {
    int n = idx >> 7, k = idx & (H - 1);
    sWhh0[n][k] = (__bf16)w_hh0[idx];
  }
  for (int idx = tid; idx < MLPD * H; idx += NTHREADS) {
    int n = idx >> 7, k = idx & (H - 1);
    sW1h[n][k] = (__bf16)w1[n * (H + IN) + k];
  }
  for (int idx = tid; idx < MLPD * 32; idx += NTHREADS) {
    int n = idx >> 5, k = idx & 31;
    sW1x[n][k] = (k < IN) ? (__bf16)w1[n * (H + IN) + H + k] : (__bf16)0.0f;
  }
  for (int idx = tid; idx < G3; idx += NTHREADS) {
    sBias[0][idx] = b_ih0[idx]; sBias[1][idx] = b_hh0[idx];
    sBias[2][idx] = b_ih1[idx]; sBias[3][idx] = b_hh1[idx];
  }
  for (int idx = tid; idx < MLPD; idx += NTHREADS) { sB1v[idx] = b1[idx]; sW2v[idx] = w2[idx]; }
  for (int idx = tid; idx < 2 * MROWS * 32; idx += NTHREADS) (&sAxt2[0][0][0])[idx] = (__bf16)0.0f;
  for (int idx = tid; idx < 2 * MROWS * H; idx += NTHREADS) {
    (&sAh[0][0][0])[idx] = (__bf16)0.0f;
    (&sH[0][0][0])[idx]  = 0.0f;
  }

  // Layer-1 weights -> persistent per-wave VGPR B-fragments (64 VGPRs/wave).
  const int wcol = n0 + l16;          // gate output column owned by this lane
  v16bf Bih1[4], Bhh1[4];
#pragma unroll
  for (int kc = 0; kc < 4; ++kc) {
    Bih1[kc] = make_bfrag_global(w_ih1, H, wcol, kc * 32 + hf * 16);
    Bhh1[kc] = make_bfrag_global(w_hh1, H, wcol, kc * 32 + hf * 16);
  }

  float met  = init_met[0];
  const float b2v = b2[0];

  __syncthreads();
  // Stage x_0 into ping-pong buffer 0 (after the zero-fill barrier).
  if (tid < MROWS * IN) {
    int m = tid >> 4, k = tid & 15;
    sAxt2[0][m][k] = (__bf16)x[(size_t)(b0 + m) * SEQ * IN + k];
  }
  __syncthreads();

  // ---------------- sequential scan over time ----------------
  for (int t = 0; t < SEQ; ++t) {
    const int par = t & 1;
    // Prefetch x_{t+1} (global_prefetch_b8: no counter, no wait). Same
    // threads + same address expression as the s5 staging load so the
    // address computation is shared (CSE) and stays off waves 0-3.
    if (wv >= 4 && wv < 12 && t + 1 < SEQ) {
      int q = tid - 128, m = q >> 4, k = q & 15;
      __builtin_prefetch(&x[(size_t)(b0 + m) * SEQ * IN + (size_t)(t + 1) * IN + k], 0, 3);
    }

    // s1: layer-0 gate GEMMs  Gi = x·Wih0^T,  Gh = h0·Whh0^T  (+ biases)
    {
      v8f ai = {}, ah = {};
      {
        v16bf a = load_afrag(&sAxt2[par][0][0], 32, lane, 0);
        v16bf b = load_bfrag(&sWih0[0][0], 32, lane, n0, 0);
        ai = WMMA_BF16(a, b, ai);
      }
#pragma unroll
      for (int kc = 0; kc < 4; ++kc) {
        v16bf a = load_afrag(&sAh[0][0][0], H, lane, kc * 32);
        v16bf b = load_bfrag(&sWhh0[0][0], H, lane, n0, kc * 32);
        ah = WMMA_BF16(a, b, ah);
      }
      const float bi = sBias[0][n0 + l16];
      const float bh = sBias[1][n0 + l16];
#pragma unroll
      for (int r = 0; r < 8; ++r) {
        sG[0][r + 8 * hf][n0 + l16] = ai[r] + bi;
        sG[1][r + 8 * hf][n0 + l16] = ah[r] + bh;
      }
    }
    __syncthreads();

    // s2: h0 update (vectorized: 512 threads x 4 units, b128 LDS traffic)
    gru_update(&sG[0][0][0], &sG[1][0][0], &sH[0][0][0], &sAh[0][0][0], tid);
    __syncthreads();

    // s3: layer-1 gate GEMMs  Gi = h0_new·Wih1^T,  Gh = h1·Whh1^T  (B in VGPRs)
    {
      v8f ai = {}, ah = {};
#pragma unroll
      for (int kc = 0; kc < 4; ++kc) {
        v16bf a0 = load_afrag(&sAh[0][0][0], H, lane, kc * 32);
        ai = WMMA_BF16(a0, Bih1[kc], ai);
        v16bf a1 = load_afrag(&sAh[1][0][0], H, lane, kc * 32);
        ah = WMMA_BF16(a1, Bhh1[kc], ah);
      }
      const float bi = sBias[2][n0 + l16];
      const float bh = sBias[3][n0 + l16];
#pragma unroll
      for (int r = 0; r < 8; ++r) {
        sG[0][r + 8 * hf][n0 + l16] = ai[r] + bi;
        sG[1][r + 8 * hf][n0 + l16] = ah[r] + bh;
      }
    }
    __syncthreads();

    // s4: h1 update
    gru_update(&sG[0][0][0], &sG[1][0][0], &sH[1][0][0], &sAh[1][0][0], tid);
    __syncthreads();

    // s5: head layer 1 on waves 0-3; concurrently, waves 4-11 stage x_{t+1}
    // into the other ping-pong panel (their load-wait hides behind the WMMAs).
    if (wv < 4) {
      v8f acc = {};
#pragma unroll
      for (int kc = 0; kc < 4; ++kc) {
        v16bf a = load_afrag(&sAh[1][0][0], H, lane, kc * 32);
        v16bf b = load_bfrag(&sW1h[0][0], H, lane, n0, kc * 32);
        acc = WMMA_BF16(a, b, acc);
      }
      {
        v16bf a = load_afrag(&sAxt2[par][0][0], 32, lane, 0);
        v16bf b = load_bfrag(&sW1x[0][0], 32, lane, n0, 0);
        acc = WMMA_BF16(a, b, acc);
      }
      const float bb = sB1v[n0 + l16];
#pragma unroll
      for (int r = 0; r < 8; ++r) {
        float v = acc[r] + bb;
        sHdn[r + 8 * hf][n0 + l16] = (v > 0.0f) ? v : 0.0f;
      }
    } else if (wv < 12 && t + 1 < SEQ) {
      int q = tid - 128, m = q >> 4, k = q & 15;
      sAxt2[par ^ 1][m][k] =
          (__bf16)x[(size_t)(b0 + m) * SEQ * IN + (size_t)(t + 1) * IN + k];
    }
    __syncthreads();

    // s6: head layer 2 + tanh*0.125 + running cumsum (wave 0, lanes 0..15)
    if (tid < MROWS) {
      float s = b2v;
      const float4* hp = (const float4*)(&sHdn[tid][0]);
      const float4* wp = (const float4*)(&sW2v[0]);
#pragma unroll
      for (int j = 0; j < MLPD / 4; ++j) {
        float4 hv = hp[j], wq = wp[j];
        s += hv.x * wq.x + hv.y * wq.y + hv.z * wq.z + hv.w * wq.w;
      }
      met += tanh_fast(s) * 0.125f;
      out[(size_t)(b0 + tid) * SEQ + t] = met;
    }
    // No top-of-loop barrier needed: the s5->s6 barrier orders the x staging
    // store against next step's s1 read, and s6's sHdn read is protected by
    // the s1->s2 ... s4->s5 barriers before sHdn is rewritten.
  }
}

extern "C" void kernel_launch(void* const* d_in, const int* in_sizes, int n_in,
                              void* d_out, int out_size, void* d_ws, size_t ws_size,
                              hipStream_t stream) {
  (void)in_sizes; (void)n_in; (void)out_size; (void)d_ws; (void)ws_size;
  const float* x      = (const float*)d_in[0];
  const float* w_ih0  = (const float*)d_in[1];
  const float* w_hh0  = (const float*)d_in[2];
  const float* b_ih0  = (const float*)d_in[3];
  const float* b_hh0  = (const float*)d_in[4];
  const float* w_ih1  = (const float*)d_in[5];
  const float* w_hh1  = (const float*)d_in[6];
  const float* b_ih1  = (const float*)d_in[7];
  const float* b_hh1  = (const float*)d_in[8];
  const float* w1     = (const float*)d_in[9];
  const float* b1     = (const float*)d_in[10];
  const float* w2     = (const float*)d_in[11];
  const float* b2     = (const float*)d_in[12];
  const float* im     = (const float*)d_in[13];
  float* out = (float*)d_out;

  gru_metabolism_kernel<<<dim3(BATCH / MROWS), dim3(NTHREADS), 0, stream>>>(
      x, w_ih0, w_hh0, b_ih0, b_hh0, w_ih1, w_hh1, b_ih1, b_hh1,
      w1, b1, w2, b2, im, out);
}
